// FiveLayers_58325655880088
// MI455X (gfx1250) — matrix operand order repaired and checked by reading, not compiled, not run on hardware
//
#include <hip/hip_runtime.h>
#include <hip/hip_bf16.h>

// ---------------- CDNA5 WMMA types ----------------
typedef __bf16 v16bf __attribute__((ext_vector_type(16)));
typedef float  v8f   __attribute__((ext_vector_type(8)));
typedef short  v16s  __attribute__((ext_vector_type(16)));

union V16U {
    v16s  s;
    v16bf b;
    unsigned short us[16];
    struct { uint4 lo, hi; } q;
};

__device__ __forceinline__ unsigned short f2bf(float f) {
    union { __bf16 b; unsigned short u; } cv;
    cv.b = (__bf16)f;
    return cv.u;
}

__device__ __forceinline__ v8f wmma_bf16(v16bf a, v16bf b, v8f c) {
    return __builtin_amdgcn_wmma_f32_16x16x32_bf16(
        false, a, false, b, (short)0, c, false, false);
}

#define TT 2048
#define BB 32
#define II 64
#define HH 256
#define GG 1024   // 4*H
#define OO 64

// LDS padded row strides (64-bank conflict avoidance)
#define GP  1028  // gates row stride in f32:  8*1028 % 64 == 32  -> halves split banks
#define HP  264   // h row stride in bf16: lane ln -> bank 4*ln (b128 conflict-free)

// ---------------------------------------------------------------------------
// One-shot f32 -> bf16 conversions.
// ---------------------------------------------------------------------------
__global__ __launch_bounds__(256) void cvt_bf16(const float* __restrict__ in,
                                                unsigned short* __restrict__ out,
                                                int n) {
    int i = blockIdx.x * 256 + threadIdx.x;
    if (i < n) out[i] = f2bf(in[i]);
}

// Layer-1 input: x [B, T, I] f32 -> a [T*B, I] bf16 with row r = t*32 + b.
__global__ __launch_bounds__(256) void cvt_x_transpose(const float* __restrict__ x,
                                                       unsigned short* __restrict__ out) {
    int g = blockIdx.x * 256 + threadIdx.x;      // 65536 * 16 quads
    int r = g >> 4;
    int q = (g & 15) * 4;
    int t = r >> 5, b = r & 31;
    float4 v = *(const float4*)(x + (size_t)b * (TT * II) + (size_t)t * II + q);
    unsigned short* o = out + (size_t)r * II + q;
    o[0] = f2bf(v.x); o[1] = f2bf(v.y); o[2] = f2bf(v.z); o[3] = f2bf(v.w);
}

// ---------------------------------------------------------------------------
// Strided WMMA GEMM, bf16 operands, f32 out:
//   Out[r, n] = sum_k A[r,k] * W[n,k] + bias(n),  r = t*32 + b
// 256 threads = 8 waves; each wave computes a 32(M) x 64(N) tile
// (2 M-tiles share every B fragment: 1.5 b128 loads per WMMA).
// ---------------------------------------------------------------------------
__global__ __launch_bounds__(256) void wmma_gemm_bf16(
    const unsigned short* __restrict__ A,
    const unsigned short* __restrict__ W, int K,
    const float* __restrict__ bias0, const float* __restrict__ bias1,
    float* __restrict__ Out, long long ost_t, long long ost_b)
{
    const int tid  = threadIdx.x;
    const int wave = tid >> 5;
    const int lane = tid & 31;
    const int lh   = lane >> 4;
    const int ln   = lane & 15;

    const int m_base  = blockIdx.x * 256 + wave * 32;
    const int n_block = blockIdx.y * 64;

    const unsigned short* arow0 = A + (size_t)(m_base + ln) * K;
    const unsigned short* arow1 = A + (size_t)(m_base + 16 + ln) * K;

    v8f acc[2][4];
#pragma unroll
    for (int nt = 0; nt < 4; ++nt) {
        int n = n_block + nt * 16 + ln;
        float bv = bias0[n] + (bias1 ? bias1[n] : 0.0f);
#pragma unroll
        for (int r = 0; r < 8; ++r) { acc[0][nt][r] = bv; acc[1][nt][r] = bv; }
    }

    const int kb_off = lh ? 8 : 0;

    for (int kt = 0; kt < K; kt += 32) {
        const int kbase = kt + lh * 8;
        V16U a0, a1;
        a0.q.lo = *(const uint4*)(arow0 + kbase);
        a0.q.hi = *(const uint4*)(arow0 + kbase + 16);
        a1.q.lo = *(const uint4*)(arow1 + kbase);
        a1.q.hi = *(const uint4*)(arow1 + kbase + 16);
#pragma unroll
        for (int nt = 0; nt < 4; ++nt) {
            int n = n_block + nt * 16 + ln;
            const unsigned short* wrow = W + (size_t)n * K + kt + kb_off;
            V16U bf;
            bf.q.lo = *(const uint4*)(wrow);
            bf.q.hi = *(const uint4*)(wrow + 16);
            acc[0][nt] = wmma_bf16(a0.b, bf.b, acc[0][nt]);
            acc[1][nt] = wmma_bf16(a1.b, bf.b, acc[1][nt]);
        }
    }

#pragma unroll
    for (int mt = 0; mt < 2; ++mt)
#pragma unroll
        for (int nt = 0; nt < 4; ++nt) {
            int n = n_block + nt * 16 + ln;
#pragma unroll
            for (int r = 0; r < 8; ++r) {
                int mm = m_base + mt * 16 + lh * 8 + r;
                Out[(size_t)(mm >> 5) * ost_t + (size_t)(mm & 31) * ost_b + n] = acc[mt][nt][r];
            }
        }
}

// ---------------------------------------------------------------------------
// Persistent LSTM scan: one 1024-thread workgroup (32 wave32s), T=2048 steps.
// w_hh bf16 register-resident (128 VGPRs/wave).  h in padded LDS (bf16),
// c in registers, gates in padded LDS.  xg contribution is added in the
// elementwise phase from registers prefetched one step ahead (latency fully
// hidden behind the WMMA phase), so the WMMA chain never waits on global.
// ---------------------------------------------------------------------------
__global__ __launch_bounds__(1024, 1) void lstm_scan(
    const float* __restrict__ xg,              // [T, B, 1024] f32
    const unsigned short* __restrict__ whh,    // [1024, 256] bf16
    unsigned short* __restrict__ y)            // [T, B, 256] bf16
{
    extern __shared__ char smem[];
    float*          gates = (float*)smem;                          // 32 x GP
    unsigned short* hb    = (unsigned short*)(smem + BB * GP * 4); // 32 x HP bf16

    const int tid  = threadIdx.x;
    const int wave = tid >> 5;
    const int lane = tid & 31;
    const int lh   = lane >> 4;
    const int ln   = lane & 15;
    const int n0   = wave * 32;

    // ---- register-resident w_hh B-fragments
    v16bf bw[2][8];
#pragma unroll
    for (int nt = 0; nt < 2; ++nt) {
        int n = n0 + nt * 16 + ln;
        const unsigned short* wrow = whh + (size_t)n * HH + (lh ? 8 : 0);
#pragma unroll
        for (int kt = 0; kt < 8; ++kt) {
            V16U u;
            u.q.lo = *(const uint4*)(wrow + kt * 32);
            u.q.hi = *(const uint4*)(wrow + kt * 32 + 16);
            bw[nt][kt] = u.b;
        }
    }

    // ---- h = 0 in LDS, c = 0 in registers
    for (int i = tid; i < BB * HP; i += 1024) hb[i] = 0;
    float creg[8];
#pragma unroll
    for (int q = 0; q < 8; ++q) creg[q] = 0.0f;

    // ---- prefetch xg for t = 0 (each thread: 8 (b,j) pairs x 4 gates)
    float xr[32];
#pragma unroll
    for (int q = 0; q < 8; ++q) {
        int p = q * 1024 + tid;
        const float* src = xg + ((p >> 8) * GG + (p & 255));
        xr[q * 4 + 0] = src[0];
        xr[q * 4 + 1] = src[256];
        xr[q * 4 + 2] = src[512];
        xr[q * 4 + 3] = src[768];
    }

    __syncthreads();

    for (int t = 0; t < TT; ++t) {
        // ---- gates = h @ w_hh^T  (A from LDS, B from VGPRs, C = 0)
        v8f acc[2][2];
#pragma unroll
        for (int mt = 0; mt < 2; ++mt)
#pragma unroll
            for (int nt = 0; nt < 2; ++nt)
#pragma unroll
                for (int r = 0; r < 8; ++r) acc[mt][nt][r] = 0.0f;

#pragma unroll
        for (int kt = 0; kt < 8; ++kt) {
            const int kbase = kt * 32 + lh * 8;
            V16U a0, a1;
            a0.q.lo = *(const uint4*)(hb + ln * HP + kbase);
            a0.q.hi = *(const uint4*)(hb + ln * HP + kbase + 16);
            a1.q.lo = *(const uint4*)(hb + (16 + ln) * HP + kbase);
            a1.q.hi = *(const uint4*)(hb + (16 + ln) * HP + kbase + 16);
#pragma unroll
            for (int nt = 0; nt < 2; ++nt) {
                acc[0][nt] = wmma_bf16(a0.b, bw[nt][kt], acc[0][nt]);
                acc[1][nt] = wmma_bf16(a1.b, bw[nt][kt], acc[1][nt]);
            }
        }

        // ---- spill gate pre-activations to LDS
#pragma unroll
        for (int mt = 0; mt < 2; ++mt)
#pragma unroll
            for (int nt = 0; nt < 2; ++nt) {
                int n = n0 + nt * 16 + ln;
#pragma unroll
                for (int r = 0; r < 8; ++r) {
                    int mm = mt * 16 + lh * 8 + r;
                    gates[mm * GP + n] = acc[mt][nt][r];
                }
            }
        __syncthreads();

        // ---- elementwise gate math (xg added here, from prefetched regs)
#pragma unroll
        for (int q = 0; q < 8; ++q) {
            int p = q * 1024 + tid;
            int b = p >> 8;
            int j = p & 255;
            float gi = gates[b * GP +        j] + xr[q * 4 + 0];
            float gf = gates[b * GP + 256 +  j] + xr[q * 4 + 1];
            float gg = gates[b * GP + 512 +  j] + xr[q * 4 + 2];
            float go = gates[b * GP + 768 +  j] + xr[q * 4 + 3];
            gi = 1.0f / (1.0f + __expf(-gi));
            gf = 1.0f / (1.0f + __expf(-gf));
            gg = 1.0f - 2.0f / (1.0f + __expf(2.0f * gg));   // tanh
            go = 1.0f / (1.0f + __expf(-go));
            float c = gf * creg[q] + gi * gg;
            creg[q] = c;
            float th = 1.0f - 2.0f / (1.0f + __expf(2.0f * c));
            unsigned short hbits = f2bf(go * th);
            hb[b * HP + j] = hbits;
            y[(size_t)t * (BB * HH) + p] = hbits;   // p == b*256 + j
        }

        // ---- prefetch next step's xg (consumed 2 barriers + 1 WMMA phase later)
        if (t + 1 < TT) {
            const float* xgt = xg + (size_t)(t + 1) * (BB * GG);
#pragma unroll
            for (int q = 0; q < 8; ++q) {
                int p = q * 1024 + tid;
                const float* src = xgt + ((p >> 8) * GG + (p & 255));
                xr[q * 4 + 0] = src[0];
                xr[q * 4 + 1] = src[256];
                xr[q * 4 + 2] = src[512];
                xr[q * 4 + 3] = src[768];
            }
        }
        __syncthreads();
    }
}

// ---------------------------------------------------------------------------
extern "C" void kernel_launch(void* const* d_in, const int* in_sizes, int n_in,
                              void* d_out, int out_size, void* d_ws, size_t ws_size,
                              hipStream_t stream) {
    (void)in_sizes; (void)n_in; (void)out_size; (void)ws_size;

    const float* x    = (const float*)d_in[0];
    const float* fc_w = (const float*)d_in[21];
    const float* fc_b = (const float*)d_in[22];

    // workspace layout
    float*          xg    = (float*)d_ws;                                  // 256 MB
    unsigned short* ybf   = (unsigned short*)(xg + (size_t)TT * BB * GG);  // 32 MB
    unsigned short* a1bf  = ybf  + (size_t)TT * BB * HH;                   // 8 MB
    unsigned short* wihbf = a1bf + (size_t)TT * BB * II;
    unsigned short* whhbf = wihbf + (size_t)GG * HH;

    const size_t scan_lds = (size_t)BB * GP * 4 + (size_t)BB * HP * 2;  // 148480 B
    (void)hipFuncSetAttribute((const void*)lstm_scan,
                              hipFuncAttributeMaxDynamicSharedMemorySize,
                              (int)scan_lds);

    const dim3 gemm_blk(256);
    const dim3 xg_grid(256, 16);   // 65536/256 x 1024/64

    cvt_x_transpose<<<(TT * BB * 16) / 256, 256, 0, stream>>>(x, a1bf);

    for (int li = 0; li < 5; ++li) {
        const float* w_ih = (const float*)d_in[1 + 4 * li + 0];
        const float* w_hh = (const float*)d_in[1 + 4 * li + 1];
        const float* b_ih = (const float*)d_in[1 + 4 * li + 2];
        const float* b_hh = (const float*)d_in[1 + 4 * li + 3];
        const int K = (li == 0) ? II : HH;

        cvt_bf16<<<(GG * K + 255) / 256, 256, 0, stream>>>(w_ih, wihbf, GG * K);
        cvt_bf16<<<(GG * HH + 255) / 256, 256, 0, stream>>>(w_hh, whhbf, GG * HH);

        const unsigned short* Abf = (li == 0) ? a1bf : ybf;
        wmma_gemm_bf16<<<xg_grid, gemm_blk, 0, stream>>>(
            Abf, wihbf, K, b_ih, b_hh,
            xg, (long long)BB * GG, (long long)GG);

        lstm_scan<<<1, 1024, scan_lds, stream>>>(xg, whhbf, ybf);
    }

    // final FC: out[b, t, o] = y5[t, b, :] . fc_w[o, :] + fc_b[o]
    cvt_bf16<<<(OO * HH + 255) / 256, 256, 0, stream>>>(fc_w, wihbf, OO * HH);
    wmma_gemm_bf16<<<dim3(256, 1), gemm_blk, 0, stream>>>(
        ybf, wihbf, HH, fc_b, nullptr,
        (float*)d_out, (long long)OO, (long long)TT * OO);
}
